// Attention_2963527434392
// MI455X (gfx1250) — matrix operand order repaired and checked by reading, not compiled, not run on hardware
//
#include <hip/hip_runtime.h>

// ---------------------------------------------------------------------------
// Bahdanau attention for MI455X (gfx1250, wave32, WMMA).
//   B=32, S=2048, H=512, 2H=1024.
//   Dominant GEMM (enc @ We, 68.7 GFLOP) runs on v_wmma_f32_16x16x32_bf16
//   (bf16 inputs, f32 accumulate) -> lands at the HBM roofline (~512MB moved).
// ---------------------------------------------------------------------------

#define Hh   512
#define TWOH 1024
#define Bb   32
#define Ss   2048

typedef __attribute__((ext_vector_type(16))) __bf16 v16bf;
typedef __attribute__((ext_vector_type(8)))  float  v8f;

union Frag {
    v16bf v;
    uint4 q[2];
};

__device__ __forceinline__ unsigned short f2bf(float f) {
    // round-to-nearest-even f32 -> bf16
    unsigned int u = __float_as_uint(f);
    unsigned int r = u + 0x7FFFu + ((u >> 16) & 1u);
    return (unsigned short)(r >> 16);
}

// ---------------------------------------------------------------------------
// Kernel 1: hWh[b][h] = sum_k hidden[b][k] * Wa[k][h] + ba[h]   (32x512x512)
// ---------------------------------------------------------------------------
__global__ void __launch_bounds__(256) hwh_kernel(const float* __restrict__ hidden,
                                                  const float* __restrict__ Wa,
                                                  const float* __restrict__ ba,
                                                  float* __restrict__ hwhb) {
    __shared__ float hs[Hh];
    const int b = blockIdx.x;
    const int tid = threadIdx.x;
    for (int i = tid; i < Hh; i += 256) hs[i] = hidden[b * Hh + i];
    __syncthreads();
    for (int h = tid; h < Hh; h += 256) {
        float acc = ba[h];
        for (int k = 0; k < Hh; ++k) acc += hs[k] * Wa[(size_t)k * Hh + h];
        hwhb[b * Hh + h] = acc;
    }
}

// ---------------------------------------------------------------------------
// Kernel 2: WeT[h][k] = bf16(Wa[H + k][h])   (transposed so WMMA B-fragments
// are contiguous 128-bit loads per lane).  32x32 LDS tile transpose.
// ---------------------------------------------------------------------------
__global__ void __launch_bounds__(256) wet_kernel(const float* __restrict__ Wa,
                                                  unsigned short* __restrict__ WeT) {
    __shared__ float t[32][33];
    const int k0 = blockIdx.x * 32;   // 32 k-tiles
    const int h0 = blockIdx.y * 32;   // 16 h-tiles
    const int i = threadIdx.x >> 5;   // 0..7
    const int j = threadIdx.x & 31;
    for (int ii = i; ii < 32; ii += 8)
        t[ii][j] = Wa[(size_t)(Hh + k0 + ii) * Hh + h0 + j];
    __syncthreads();
    for (int ii = i; ii < 32; ii += 8)
        WeT[(size_t)(h0 + ii) * TWOH + k0 + j] = f2bf(t[j][ii]);
}

// ---------------------------------------------------------------------------
// Kernel 3: fused  scores[b][s] = v . tanh( enc[b] @ We + hWh[b] + ba )
// Block = (32 S-rows, all 512 H-cols) of one batch.  8 waves, each wave owns
// 64 H-columns -> 2 row-tiles x 4 col-tiles of 16x16 f32 accumulators.
// K = 1024, chunked by 128 through LDS (f32 -> bf16 conversion on the fly).
// 256 v_wmma_f32_16x16x32_bf16 per wave.
// ---------------------------------------------------------------------------
__global__ void __launch_bounds__(256) score_kernel(const float* __restrict__ enc,
                                                    const unsigned short* __restrict__ WeT,
                                                    const float* __restrict__ hwhb,
                                                    const float* __restrict__ vvec,
                                                    float* __restrict__ scores) {
    __shared__ __align__(16) unsigned short Atile[32 * 136];  // 128 + 8 pad (bank skew)
    __shared__ float spart[8][32];

    const int b   = blockIdx.y;
    const int s0  = blockIdx.x * 32;
    const int tid = threadIdx.x;
    const int lane = tid & 31;
    const int w    = tid >> 5;     // wave 0..7
    const int l15  = lane & 15;
    const int hi   = lane >> 4;    // lane half (K-split in A/B fragments)

    const float* Abase = enc + ((size_t)b * Ss + s0) * TWOH;

    v8f acc[2][4] = {};

    // staging role: 8 threads per row, 16 floats each
    const int sm = tid >> 3;          // row 0..31
    const int sk = (tid & 7) * 16;    // col start within 128-chunk

    for (int kc = 0; kc < 8; ++kc) {
        // ---- stage A chunk (32 x 128 f32 -> bf16) into LDS ----
        const float4* rp = (const float4*)(Abase + (size_t)sm * TWOH + kc * 128 + sk);
#pragma unroll
        for (int jj = 0; jj < 4; ++jj) {
            float4 f = rp[jj];
            uint2 pk;
            pk.x = (unsigned)f2bf(f.x) | ((unsigned)f2bf(f.y) << 16);
            pk.y = (unsigned)f2bf(f.z) | ((unsigned)f2bf(f.w) << 16);
            *(uint2*)&Atile[sm * 136 + sk + jj * 4] = pk;
        }
        __syncthreads();

#pragma unroll
        for (int kk = 0; kk < 4; ++kk) {
            // A fragments (contiguous K: lanes 0-15: K 0-15, lanes 16-31: K 16-31)
            const int kb = kk * 32 + hi * 16;
            Frag a0, a1;
            const unsigned short* ap0 = &Atile[l15 * 136 + kb];
            a0.q[0] = *(const uint4*)ap0;
            a0.q[1] = *(const uint4*)(ap0 + 8);
            const unsigned short* ap1 = &Atile[(16 + l15) * 136 + kb];
            a1.q[0] = *(const uint4*)ap1;
            a1.q[1] = *(const uint4*)(ap1 + 8);

            const int kg = kc * 128 + kk * 32 + hi * 16;
#pragma unroll
            for (int c = 0; c < 4; ++c) {
                // B fragment (lanes 0-15: K 0-15, lanes 16-31: K 16-31, N = l15)
                const int n = w * 64 + c * 16 + l15;
                const uint4* bp = (const uint4*)(WeT + (size_t)n * TWOH + kg);
                Frag bf;
                bf.q[0] = bp[0];
                bf.q[1] = bp[1];
                acc[0][c] = __builtin_amdgcn_wmma_f32_16x16x32_bf16(
                    false, a0.v, false, bf.v, (short)0, acc[0][c], false, false);
                acc[1][c] = __builtin_amdgcn_wmma_f32_16x16x32_bf16(
                    false, a1.v, false, bf.v, (short)0, acc[1][c], false, false);
            }
        }
        __syncthreads();
    }

    // ---- fused epilogue: +hWh+ba, tanh, *v, reduce over H ----
    float rowsum[2][8];
#pragma unroll
    for (int t = 0; t < 2; ++t)
#pragma unroll
        for (int r = 0; r < 8; ++r) rowsum[t][r] = 0.0f;

#pragma unroll
    for (int c = 0; c < 4; ++c) {
        const int n = w * 64 + c * 16 + l15;           // C/D layout: N = lane & 15
        const float hw = hwhb[b * Hh + n];
        const float vv = vvec[n];
#pragma unroll
        for (int t = 0; t < 2; ++t)
#pragma unroll
            for (int r = 0; r < 8; ++r) {              // C/D: M = r + 8*hi (+16*t)
                float e = tanhf(acc[t][c][r] + hw);
                rowsum[t][r] += e * vv;
            }
    }
    // reduce across the 16 lanes holding the 16 N-values of each row
#pragma unroll
    for (int t = 0; t < 2; ++t)
#pragma unroll
        for (int r = 0; r < 8; ++r) {
            float x = rowsum[t][r];
            x += __shfl_xor(x, 1);
            x += __shfl_xor(x, 2);
            x += __shfl_xor(x, 4);
            x += __shfl_xor(x, 8);
            rowsum[t][r] = x;
        }
    if (l15 == 0) {
#pragma unroll
        for (int t = 0; t < 2; ++t)
#pragma unroll
            for (int r = 0; r < 8; ++r)
                spart[w][t * 16 + hi * 8 + r] = rowsum[t][r];
    }
    __syncthreads();
    if (tid < 32) {
        float s = 0.0f;
#pragma unroll
        for (int q = 0; q < 8; ++q) s += spart[q][tid];
        scores[(size_t)b * Ss + s0 + tid] = s;
    }
}

// ---------------------------------------------------------------------------
// Kernel 4: softmax over S=2048, in place, one block per batch row.
// ---------------------------------------------------------------------------
__global__ void __launch_bounds__(256) softmax_kernel(float* __restrict__ attn) {
    __shared__ float red[256];
    const int b = blockIdx.x;
    const int tid = threadIdx.x;
    float* row = attn + (size_t)b * Ss;
    float vals[8];
    float mx = -3.402823466e38f;
#pragma unroll
    for (int i = 0; i < 8; ++i) {
        vals[i] = row[tid + 256 * i];
        mx = fmaxf(mx, vals[i]);
    }
    red[tid] = mx;
    __syncthreads();
    for (int off = 128; off > 0; off >>= 1) {
        if (tid < off) red[tid] = fmaxf(red[tid], red[tid + off]);
        __syncthreads();
    }
    mx = red[0];
    __syncthreads();
    float sum = 0.0f;
#pragma unroll
    for (int i = 0; i < 8; ++i) {
        vals[i] = __expf(vals[i] - mx);
        sum += vals[i];
    }
    red[tid] = sum;
    __syncthreads();
    for (int off = 128; off > 0; off >>= 1) {
        if (tid < off) red[tid] += red[tid + off];
        __syncthreads();
    }
    const float inv = 1.0f / red[0];
#pragma unroll
    for (int i = 0; i < 8; ++i) row[tid + 256 * i] = vals[i] * inv;
}

// ---------------------------------------------------------------------------
// Kernel 5: deterministic partial context: partial[i][b][e] over S-slice i.
// ---------------------------------------------------------------------------
__global__ void __launch_bounds__(256) context_partial_kernel(const float* __restrict__ enc,
                                                              const float* __restrict__ attn,
                                                              float* __restrict__ partial) {
    __shared__ float at[256];
    const int i = blockIdx.x;   // 0..7 S-slice
    const int b = blockIdx.y;
    const int tid = threadIdx.x;
    at[tid] = attn[(size_t)b * Ss + i * 256 + tid];
    __syncthreads();
    const float4* base = (const float4*)(enc + ((size_t)b * Ss + i * 256) * TWOH);
    float4 acc = make_float4(0.f, 0.f, 0.f, 0.f);
    for (int s = 0; s < 256; ++s) {
        const float wgt = at[s];
        float4 x = base[(size_t)s * 256 + tid];
        acc.x += wgt * x.x;
        acc.y += wgt * x.y;
        acc.z += wgt * x.z;
        acc.w += wgt * x.w;
    }
    ((float4*)(partial + ((size_t)i * Bb + b) * TWOH))[tid] = acc;
}

// ---------------------------------------------------------------------------
// Kernel 6: reduce 8 partials -> context (32 x 1024).
// ---------------------------------------------------------------------------
__global__ void __launch_bounds__(256) context_reduce_kernel(const float* __restrict__ partial,
                                                             float* __restrict__ ctx) {
    const int gid = blockIdx.x * 256 + threadIdx.x;   // < 32768
    const int b = gid >> 10;
    const int e = gid & 1023;
    float s = 0.0f;
#pragma unroll
    for (int i = 0; i < 8; ++i) s += partial[((size_t)i * Bb + b) * TWOH + e];
    ctx[gid] = s;
}

// ---------------------------------------------------------------------------
extern "C" void kernel_launch(void* const* d_in, const int* in_sizes, int n_in,
                              void* d_out, int out_size, void* d_ws, size_t ws_size,
                              hipStream_t stream) {
    (void)in_sizes; (void)n_in; (void)out_size; (void)ws_size;
    const float* hidden = (const float*)d_in[0];   // (1, 32, 512)
    const float* enc    = (const float*)d_in[1];   // (32, 2048, 1024)
    const float* Wa     = (const float*)d_in[2];   // (1536, 512)
    const float* ba     = (const float*)d_in[3];   // (512,)
    const float* v      = (const float*)d_in[4];   // (512,)

    float* out  = (float*)d_out;
    float* ctx  = out;                 // 32*1024 floats
    float* attn = out + Bb * TWOH;     // 32*2048 floats (scores -> softmax in place)

    // workspace layout (~2.1 MB):
    float* ws      = (float*)d_ws;
    float* hwhb    = ws;                                   // 32*512 f32      (64 KB)
    float* partial = ws + Bb * Hh;                         // 8*32*1024 f32   (1 MB)
    unsigned short* WeT = (unsigned short*)(ws + Bb * Hh + 8 * Bb * TWOH); // 512*1024 bf16 (1 MB)

    hwh_kernel<<<Bb, 256, 0, stream>>>(hidden, Wa, ba, hwhb);
    wet_kernel<<<dim3(32, 16), 256, 0, stream>>>(Wa, WeT);
    score_kernel<<<dim3(Ss / 32, Bb), 256, 0, stream>>>(enc, WeT, hwhb, v, attn);
    softmax_kernel<<<Bb, 256, 0, stream>>>(attn);
    context_partial_kernel<<<dim3(8, Bb), 256, 0, stream>>>(enc, attn, partial);
    context_reduce_kernel<<<128, 256, 0, stream>>>(partial, ctx);
}
